// MoE_66554813219386
// MI455X (gfx1250) — compile-verified, matrix operand order checked by reading, not run on hardware
//
#include <hip/hip_runtime.h>
#include <cstdint>
#include <cstddef>

// ---------------------------------------------------------------------------
// MoE (top-2, 8 experts) for MI455X / gfx1250, wave32 + WMMA bf16.
// D=1024, H=512, E=8, ob=2, N=4096 tokens per ob, K(top)=2.
// GEMMs: 32x32 register tile per wave (2x2 fragments), A staged in LDS via
// global_load_async_to_lds_b128 (ASYNCcnt), B direct from L2-resident weights.
// ---------------------------------------------------------------------------

typedef __attribute__((ext_vector_type(16))) __bf16 v16bf;
typedef __attribute__((ext_vector_type(8)))  __bf16 v8bf;
typedef __attribute__((ext_vector_type(4)))  __bf16 v4bf;
typedef __attribute__((ext_vector_type(8)))  float  v8f;

#define OB      2
#define NTOK    4096          // tokens per outer batch (ib*slen)
#define DDIM    1024
#define HDIM    512
#define NEXP    8
#define TOPK    2
#define MPAD    8448          // 8192 routed slots + up to 8*32 padding, per ob
#define MAXT    528           // max 32-row tiles across both ob
#define EXS13   (HDIM * DDIM) // expert stride for w1/w3 (rows=H, ld=D)
#define EXS2    (DDIM * HDIM) // expert stride for w2   (rows=D, ld=H)

#define KCHUNK   512                    // K columns staged in LDS per pass
#define LDSROW   ((KCHUNK + 8) * 2)     // 1040 B/row: 4-bank rotation, no conflicts
#define LDSBYTES (32 * LDSROW)          // 33280 B

// ---------------------------------------------------------------------------
// fp32 -> bf16 elementwise conversion
// ---------------------------------------------------------------------------
__global__ void moe_cvt_bf16(const float* __restrict__ in,
                             __bf16* __restrict__ out, int n) {
    int i = blockIdx.x * 256 + threadIdx.x;
    if (i < n) out[i] = (__bf16)in[i];
}

// ---------------------------------------------------------------------------
// init: zero counters/cursors, tok_map = -1
// ---------------------------------------------------------------------------
__global__ void moe_init(int* __restrict__ counts, int* __restrict__ cursors,
                         int* __restrict__ tok_map, int ntm) {
    int i = blockIdx.x * 256 + threadIdx.x;
    if (i < OB * NEXP) { counts[i] = 0; cursors[i] = 0; }
    if (i < ntm) tok_map[i] = -1;
}

// ---------------------------------------------------------------------------
// router: one wave32 per token. scores = sigmoid(x . gate_w[e]); top-2 with
// expert bias tiebreak (lowest index wins ties, matching jax top_k).
// ---------------------------------------------------------------------------
__global__ void moe_router(const float* __restrict__ x,
                           const float* __restrict__ gate_w,
                           const float* __restrict__ expert_bias,
                           int* __restrict__ idx2, float* __restrict__ score2,
                           int* __restrict__ counts) {
    const int wave  = threadIdx.x >> 5;
    const int lane  = threadIdx.x & 31;
    const int token = blockIdx.x * 4 + wave;          // [0, OB*NTOK)
    const float* xr = x + (size_t)token * DDIM;

    float acc[NEXP];
#pragma unroll
    for (int e = 0; e < NEXP; ++e) acc[e] = 0.0f;

    for (int j = lane; j < DDIM; j += 32) {
        float xv = xr[j];
#pragma unroll
        for (int e = 0; e < NEXP; ++e) acc[e] += xv * gate_w[e * DDIM + j];
    }
#pragma unroll
    for (int off = 16; off > 0; off >>= 1) {
#pragma unroll
        for (int e = 0; e < NEXP; ++e) acc[e] += __shfl_xor(acc[e], off, 32);
    }

    if (lane == 0) {
        const int b = token / NTOK;
        float sc[NEXP];
        int   i0 = 0;
        float b0 = -1e30f;
#pragma unroll
        for (int e = 0; e < NEXP; ++e) {
            float s = 1.0f / (1.0f + __expf(-acc[e]));
            sc[e] = s;
            float bb = s + expert_bias[e];
            if (bb > b0) { b0 = bb; i0 = e; }
        }
        int   i1 = (i0 == 0) ? 1 : 0;
        float b1 = -1e30f;
#pragma unroll
        for (int e = 0; e < NEXP; ++e) {
            if (e == i0) continue;
            float bb = sc[e] + expert_bias[e];
            if (bb > b1) { b1 = bb; i1 = e; }
        }
        idx2[token * 2 + 0]   = i0;
        idx2[token * 2 + 1]   = i1;
        score2[token * 2 + 0] = sc[i0];   // ROUTE_SCALE == 1.0
        score2[token * 2 + 1] = sc[i1];
        atomicAdd(&counts[b * NEXP + i0], 1);
        atomicAdd(&counts[b * NEXP + i1], 1);
    }
}

// ---------------------------------------------------------------------------
// plan: pad each (ob,expert) group to a multiple of 32 rows, compute offsets,
// and emit the 32-row tile table {global_row_start, expert, ob, valid}.
// ---------------------------------------------------------------------------
__global__ void moe_plan(const int* __restrict__ counts,
                         int* __restrict__ padded_off,
                         int4* __restrict__ meta) {
    if (threadIdx.x != 0 || blockIdx.x != 0) return;
    int t = 0;
    for (int b = 0; b < OB; ++b) {
        int off = 0;
        for (int e = 0; e < NEXP; ++e) {
            int c  = counts[b * NEXP + e];
            int pc = (c + 31) & ~31;
            padded_off[b * NEXP + e] = off;
            for (int i = 0; i < pc / 32; ++i)
                meta[t++] = make_int4(b * MPAD + off + i * 32, e, b, 1);
            off += pc;
        }
    }
    for (; t < MAXT; ++t) meta[t] = make_int4(0, 0, 0, 0);
}

// ---------------------------------------------------------------------------
// scatter: place each (token,k) slot into its expert region
// ---------------------------------------------------------------------------
__global__ void moe_scatter(const int* __restrict__ idx2,
                            const float* __restrict__ score2,
                            const int* __restrict__ padded_off,
                            int* __restrict__ cursors,
                            int* __restrict__ tok_map,
                            float* __restrict__ score_map) {
    int t = blockIdx.x * 256 + threadIdx.x;
    if (t >= OB * NTOK) return;
    int b = t / NTOK;
#pragma unroll
    for (int k = 0; k < TOPK; ++k) {
        int   e   = idx2[t * 2 + k];
        float s   = score2[t * 2 + k];
        int   pos = padded_off[b * NEXP + e] + atomicAdd(&cursors[b * NEXP + e], 1);
        int   g   = b * MPAD + pos;
        tok_map[g]   = t - b * NTOK;
        score_map[g] = s;
    }
}

// ---------------------------------------------------------------------------
// gather: lhs[g][:] = bf16(x[tok][:] * score); padding rows -> zeros
// ---------------------------------------------------------------------------
__global__ void moe_gather(const float* __restrict__ x,
                           const int* __restrict__ tok_map,
                           const float* __restrict__ score_map,
                           __bf16* __restrict__ lhs) {
    const int g   = blockIdx.x;                 // [0, OB*MPAD)
    const int b   = g / MPAD;
    const int tok = tok_map[g];
    const int d0  = threadIdx.x * 4;
    v4bf out;
    if (tok < 0) {
        out[0] = (__bf16)0.0f; out[1] = (__bf16)0.0f;
        out[2] = (__bf16)0.0f; out[3] = (__bf16)0.0f;
    } else {
        float s = score_map[g];
        const float4 v = *(const float4*)(x + ((size_t)(b * NTOK + tok)) * DDIM + d0);
        out[0] = (__bf16)(v.x * s); out[1] = (__bf16)(v.y * s);
        out[2] = (__bf16)(v.z * s); out[3] = (__bf16)(v.w * s);
    }
    *(v4bf*)(lhs + (size_t)g * DDIM + d0) = out;
}

// ---------------------------------------------------------------------------
// WMMA fragment load from global: row-major, K-contiguous (NT GEMM: A and B^T
// load identically). CDNA5 16-bit A layout: lane L holds row (L&15),
// k = 8*(L>>4) + {0..7} (VGPR0-3) and +16 (VGPR4-7) -> two b128 loads.
// ---------------------------------------------------------------------------
__device__ __forceinline__ v16bf moe_load_frag(const __bf16* __restrict__ p0) {
    v8bf lo = *(const v8bf*)p0;
    v8bf hi = *(const v8bf*)(p0 + 16);
    v16bf f;
#pragma unroll
    for (int i = 0; i < 8; ++i) { f[i] = lo[i]; f[i + 8] = hi[i]; }
    return f;
}

// Same fragment, sourced from the LDS-staged A tile (row stride LDSROW bytes).
__device__ __forceinline__ v16bf moe_lds_frag(const char* smem, int row, int kk) {
    const char* p = smem + row * LDSROW + kk * 2;
    v8bf lo = *(const v8bf*)p;
    v8bf hi = *(const v8bf*)(p + 32);
    v16bf f;
#pragma unroll
    for (int i = 0; i < 8; ++i) { f[i] = lo[i]; f[i + 8] = hi[i]; }
    return f;
}

__device__ __forceinline__ v8f moe_wmma_bf16(v16bf a, v16bf b, v8f c) {
    return __builtin_amdgcn_wmma_f32_16x16x32_bf16(
        /*neg_a=*/false, a, /*neg_b=*/false, b,
        /*c_mod=*/(short)0, c, /*reuse_a=*/false, /*reuse_b=*/false);
}

// ---------------------------------------------------------------------------
// Stage a 32 x KCHUNK bf16 A-tile into LDS with CDNA5 async LDS loads
// (tracked by ASYNCcnt), then fence: s_wait_asynccnt 0 + workgroup barrier.
// Generic-pointer low 32 bits are the LDS byte offset (flat aperture rule).
// ---------------------------------------------------------------------------
__device__ __forceinline__ void moe_stage_a(const __bf16* __restrict__ g0, int ldA,
                                            char* smem, int tid) {
#pragma unroll 4
    for (int c = tid; c < 32 * (KCHUNK / 8); c += 128) {   // 16B chunks
        const int row = c >> 6;            // KCHUNK/8 == 64 chunks per row
        const int cc  = c & 63;
        const __bf16* g = g0 + (size_t)row * ldA + cc * 8;
        const unsigned l = (unsigned)(uintptr_t)(smem + row * LDSROW + cc * 16);
        asm volatile("global_load_async_to_lds_b128 %0, %1, off"
                     :: "v"(l), "v"((unsigned long long)(uintptr_t)g)
                     : "memory");
    }
    asm volatile("s_wait_asynccnt 0" ::: "memory");
    __syncthreads();
}

// ---------------------------------------------------------------------------
// GEMM-13: H = silu(A . W1^T) * (A . W3^T), bf16 out.
// Block = 128 threads (4 waves). Block tile: 32 rows x 128 cols.
// Wave tile: 32x32 (2 M-frags x 2 N-frags, dual accumulators for W1/W3).
// ---------------------------------------------------------------------------
template <bool GROUPED>
__global__ __launch_bounds__(128)
void moe_gemm13(const __bf16* __restrict__ A, const __bf16* __restrict__ W1,
                const __bf16* __restrict__ W3, const int4* __restrict__ meta,
                __bf16* __restrict__ Hout, int Kdim, int ldA, int ldH) {
    __shared__ __align__(16) char smem[LDSBYTES];
    int row0, e;
    if (GROUPED) {
        int4 mt = meta[blockIdx.x];
        if (mt.w == 0) return;                 // block-uniform early exit
        row0 = mt.x; e = mt.y;
    } else {
        row0 = blockIdx.x * 32; e = 0;
    }
    const int lane = threadIdx.x & 31;
    const int wv   = threadIdx.x >> 5;
    const int lrow = lane & 15;
    const int half = lane >> 4;
    const int nb   = blockIdx.y * 128 + wv * 32;

    const __bf16* w1r0 = W1 + (size_t)e * EXS13 + (size_t)(nb + lrow) * Kdim + half * 8;
    const __bf16* w1r1 = w1r0 + (size_t)16 * Kdim;
    const __bf16* w3r0 = W3 + (size_t)e * EXS13 + (size_t)(nb + lrow) * Kdim + half * 8;
    const __bf16* w3r1 = w3r0 + (size_t)16 * Kdim;

    v8f acc1[2][2] = {{{}, {}}, {{}, {}}};
    v8f acc3[2][2] = {{{}, {}}, {{}, {}}};

    for (int ko = 0; ko < Kdim; ko += KCHUNK) {
        if (ko != 0) __syncthreads();          // protect LDS re-staging
        moe_stage_a(A + (size_t)row0 * ldA + ko, ldA, smem, threadIdx.x);
        for (int k = 0; k < KCHUNK; k += 32) {
            const int kk = k + half * 8;
            const int kg = ko + k;
            v16bf a0 = moe_lds_frag(smem, lrow,      kk);
            v16bf a1 = moe_lds_frag(smem, 16 + lrow, kk);
            v16bf b10 = moe_load_frag(w1r0 + kg);
            v16bf b11 = moe_load_frag(w1r1 + kg);
            v16bf b30 = moe_load_frag(w3r0 + kg);
            v16bf b31 = moe_load_frag(w3r1 + kg);
            acc1[0][0] = moe_wmma_bf16(a0, b10, acc1[0][0]);
            acc1[0][1] = moe_wmma_bf16(a0, b11, acc1[0][1]);
            acc1[1][0] = moe_wmma_bf16(a1, b10, acc1[1][0]);
            acc1[1][1] = moe_wmma_bf16(a1, b11, acc1[1][1]);
            acc3[0][0] = moe_wmma_bf16(a0, b30, acc3[0][0]);
            acc3[0][1] = moe_wmma_bf16(a0, b31, acc3[0][1]);
            acc3[1][0] = moe_wmma_bf16(a1, b30, acc3[1][0]);
            acc3[1][1] = moe_wmma_bf16(a1, b31, acc3[1][1]);
        }
    }
#pragma unroll
    for (int mi = 0; mi < 2; ++mi)
#pragma unroll
        for (int ni = 0; ni < 2; ++ni)
#pragma unroll
            for (int r = 0; r < 8; ++r) {
                int   m  = row0 + mi * 16 + r + half * 8;
                int   n  = nb + ni * 16 + lrow;
                float a1 = acc1[mi][ni][r];
                float hv = (a1 / (1.0f + __expf(-a1))) * acc3[mi][ni][r];
                Hout[(size_t)m * ldH + n] = (__bf16)hv;
            }
}

// ---------------------------------------------------------------------------
// GEMM-2: Out = A . W2^T (f32). GROUPED: scatter-add onto d_out via tok_map
// (fused token-combine); else direct store (shared-expert path).
// ---------------------------------------------------------------------------
template <bool GROUPED>
__global__ __launch_bounds__(128)
void moe_gemm2(const __bf16* __restrict__ A, const __bf16* __restrict__ W2,
               const int4* __restrict__ meta, const int* __restrict__ tok_map,
               float* __restrict__ Out, int Kdim, int ldA, int ldOut) {
    __shared__ __align__(16) char smem[LDSBYTES];
    int row0, e, b;
    if (GROUPED) {
        int4 mt = meta[blockIdx.x];
        if (mt.w == 0) return;
        row0 = mt.x; e = mt.y; b = mt.z;
    } else {
        row0 = blockIdx.x * 32; e = 0; b = 0;
    }
    const int lane = threadIdx.x & 31;
    const int wv   = threadIdx.x >> 5;
    const int lrow = lane & 15;
    const int half = lane >> 4;
    const int nb   = blockIdx.y * 128 + wv * 32;

    const __bf16* wr0 = W2 + (size_t)e * EXS2 + (size_t)(nb + lrow) * Kdim + half * 8;
    const __bf16* wr1 = wr0 + (size_t)16 * Kdim;

    v8f acc[2][2] = {{{}, {}}, {{}, {}}};

    for (int ko = 0; ko < Kdim; ko += KCHUNK) {
        if (ko != 0) __syncthreads();
        moe_stage_a(A + (size_t)row0 * ldA + ko, ldA, smem, threadIdx.x);
        for (int k = 0; k < KCHUNK; k += 32) {
            const int kk = k + half * 8;
            const int kg = ko + k;
            v16bf a0 = moe_lds_frag(smem, lrow,      kk);
            v16bf a1 = moe_lds_frag(smem, 16 + lrow, kk);
            v16bf b0 = moe_load_frag(wr0 + kg);
            v16bf b1 = moe_load_frag(wr1 + kg);
            acc[0][0] = moe_wmma_bf16(a0, b0, acc[0][0]);
            acc[0][1] = moe_wmma_bf16(a0, b1, acc[0][1]);
            acc[1][0] = moe_wmma_bf16(a1, b0, acc[1][0]);
            acc[1][1] = moe_wmma_bf16(a1, b1, acc[1][1]);
        }
    }
#pragma unroll
    for (int mi = 0; mi < 2; ++mi)
#pragma unroll
        for (int ni = 0; ni < 2; ++ni)
#pragma unroll
            for (int r = 0; r < 8; ++r) {
                int m = row0 + mi * 16 + r + half * 8;
                int n = nb + ni * 16 + lrow;
                if (GROUPED) {
                    int tok = tok_map[m];
                    if (tok >= 0)
                        atomicAdd(&Out[((size_t)(b * NTOK + tok)) * ldOut + n],
                                  acc[mi][ni][r]);
                } else {
                    Out[(size_t)m * ldOut + n] = acc[mi][ni][r];
                }
            }
}

// ---------------------------------------------------------------------------
// host launcher
// ---------------------------------------------------------------------------
extern "C" void kernel_launch(void* const* d_in, const int* in_sizes, int n_in,
                              void* d_out, int out_size, void* d_ws, size_t ws_size,
                              hipStream_t stream) {
    (void)in_sizes; (void)n_in; (void)out_size; (void)ws_size;
    const float* x           = (const float*)d_in[0]; // (2,2,2048,1024)
    const float* gate_w      = (const float*)d_in[1]; // (8,1024)
    const float* w1          = (const float*)d_in[2]; // (8,512,1024)
    const float* w2          = (const float*)d_in[3]; // (8,1024,512)
    const float* w3          = (const float*)d_in[4]; // (8,512,1024)
    const float* sw1         = (const float*)d_in[5]; // (512,1024)
    const float* sw2         = (const float*)d_in[6]; // (1024,512)
    const float* sw3         = (const float*)d_in[7]; // (512,1024)
    const float* expert_bias = (const float*)d_in[8]; // (8,)
    float* out = (float*)d_out;                       // (2,2,2048,1024) f32

    // ---- workspace carve-up (256B aligned) ----
    uint8_t* ws  = (uint8_t*)d_ws;
    size_t   off = 0;
    auto take = [&](size_t bytes) {
        uint8_t* p = ws + off;
        off = (off + bytes + 255) & ~(size_t)255;
        return p;
    };
    int*    counts     = (int*)   take(OB * NEXP * 4);
    int*    cursors    = (int*)   take(OB * NEXP * 4);
    int*    padded_off = (int*)   take(OB * NEXP * 4);
    int*    idx2       = (int*)   take((size_t)OB * NTOK * 2 * 4);
    float*  score2     = (float*) take((size_t)OB * NTOK * 2 * 4);
    int*    tok_map    = (int*)   take((size_t)OB * MPAD * 4);
    float*  score_map  = (float*) take((size_t)OB * MPAD * 4);
    int4*   meta       = (int4*)  take((size_t)MAXT * 16);
    __bf16* w1b        = (__bf16*)take((size_t)NEXP * EXS13 * 2);
    __bf16* w3b        = (__bf16*)take((size_t)NEXP * EXS13 * 2);
    __bf16* w2b        = (__bf16*)take((size_t)NEXP * EXS2  * 2);
    __bf16* sw1b       = (__bf16*)take((size_t)EXS13 * 2);
    __bf16* sw3b       = (__bf16*)take((size_t)EXS13 * 2);
    __bf16* sw2b       = (__bf16*)take((size_t)EXS2  * 2);
    __bf16* xb         = (__bf16*)take((size_t)OB * NTOK * DDIM * 2);
    __bf16* lhs        = (__bf16*)take((size_t)OB * MPAD * DDIM * 2);
    __bf16* hbuf       = (__bf16*)take((size_t)OB * MPAD * HDIM * 2);
    __bf16* hs         = (__bf16*)take((size_t)OB * NTOK * HDIM * 2);

    auto cvt = [&](const float* src, __bf16* dst, int n) {
        moe_cvt_bf16<<<(n + 255) / 256, 256, 0, stream>>>(src, dst, n);
    };
    // 1) precision conversion (reference casts GEMM operands to bf16)
    cvt(w1,  w1b,  NEXP * EXS13);
    cvt(w3,  w3b,  NEXP * EXS13);
    cvt(w2,  w2b,  NEXP * EXS2);
    cvt(sw1, sw1b, EXS13);
    cvt(sw3, sw3b, EXS13);
    cvt(sw2, sw2b, EXS2);
    cvt(x,   xb,   OB * NTOK * DDIM);

    // 2) init counters + tok_map
    moe_init<<<(OB * MPAD + 255) / 256, 256, 0, stream>>>(counts, cursors, tok_map,
                                                          OB * MPAD);
    // 3) router (one wave32 per token, 4 waves/block)
    moe_router<<<OB * NTOK / 4, 128, 0, stream>>>(x, gate_w, expert_bias,
                                                  idx2, score2, counts);
    // 4) group planning (padded counting-sort layout + 32-row tile table)
    moe_plan<<<1, 1, 0, stream>>>(counts, padded_off, meta);
    // 5) scatter slots, 6) gather scaled bf16 LHS
    moe_scatter<<<(OB * NTOK + 255) / 256, 256, 0, stream>>>(
        idx2, score2, padded_off, cursors, tok_map, score_map);
    moe_gather<<<OB * MPAD, 256, 0, stream>>>(x, tok_map, score_map, lhs);

    // 7) shared-expert FFN: writes base output (plain stores)
    moe_gemm13<false><<<dim3(OB * NTOK / 32, HDIM / 128), 128, 0, stream>>>(
        xb, sw1b, sw3b, nullptr, hs, DDIM, DDIM, HDIM);
    moe_gemm2<false><<<dim3(OB * NTOK / 32, DDIM / 128), 128, 0, stream>>>(
        hs, sw2b, nullptr, nullptr, out, HDIM, HDIM, DDIM);

    // 8) grouped experts: silu(A.w1^T)*(A.w3^T) then .w2^T with fused
    //    scatter-add combine onto the shared output
    moe_gemm13<true><<<dim3(MAXT, HDIM / 128), 128, 0, stream>>>(
        lhs, w1b, w3b, meta, hbuf, DDIM, DDIM, HDIM);
    moe_gemm2<true><<<dim3(MAXT, DDIM / 128), 128, 0, stream>>>(
        hbuf, w2b, meta, tok_map, out, HDIM, HDIM, DDIM);
}
